// FeatureExtraction_67482526155287
// MI455X (gfx1250) — compile-verified, matrix operand order checked by reading, not run on hardware
//
#include <hip/hip_runtime.h>
#include <hip/hip_bf16.h>

typedef __attribute__((ext_vector_type(16))) _Float16 v16h;
typedef __attribute__((ext_vector_type(8)))  _Float16 v8h;
typedef __attribute__((ext_vector_type(8)))  float    v8f;

#define ACT_NONE 0
#define ACT_RELU 1
#define ACT_SIG  2

static constexpr int PTS = 32768;   // B*N
static constexpr int NPB = 1024;    // N per batch

// ---------------- WMMA helpers (CDNA5 wave32 layouts, ISA 7.12.2) ----------

__device__ __forceinline__ v8f wmma_f16(v16h a, v16h b, v8f c) {
  return __builtin_amdgcn_wmma_f32_16x16x32_f16(
      false, a, false, b, (short)0, c, false, false);
}

__device__ __forceinline__ v8f vzero() { v8f z = {}; return z; }

// K index covered by fragment element e of lane l (same map for A rows / B cols)
__device__ __forceinline__ int frag_k_of(int l, int e) {
  int half = l >> 4;
  int v = e >> 1, par = e & 1;
  return ((v & 3) * 2) + ((v >= 4) ? 16 : 0) + half * 8 + par;
}
// inverse: element slot for K index k (lane half is (k>>3)&1)
__device__ __forceinline__ int frag_elem(int k) {
  int half = (k >> 3) & 1;
  int kp = k - 8 * half;
  int par = kp & 1;
  int kp2 = kp - par;
  int v = (kp2 < 8) ? (kp2 >> 1) : (4 + ((kp2 - 16) >> 1));
  return 2 * v + par;
}

// contiguous fragment load: lane l reads 32B at p + l*16  -> 2 x b128
__device__ __forceinline__ v16h load_frag(const _Float16* p, int lane) {
  return *(const v16h*)(p + lane * 16);
}

// async global->LDS copy of 4 bytes per lane (CDNA5, tracked by ASYNCcnt)
__device__ __forceinline__ void async_copy_b32(void* lds, const void* g) {
  asm volatile("global_load_async_to_lds_b32 %0, %1, off"
               :: "v"((unsigned)(uintptr_t)lds), "v"(g) : "memory");
}
__device__ __forceinline__ void async_wait0() {
  asm volatile("s_wait_asynccnt 0" ::: "memory");
}

// element-wise A fragment (for global f32 activations; pairs merge to b64)
template <typename F>
__device__ __forceinline__ v16h load_a_frag(F&& f, int lane) {
  int row = lane & 15, half = lane >> 4;
  v16h a;
#pragma unroll
  for (int v = 0; v < 8; ++v) {
    int kOff = ((v & 3) * 2) + ((v >= 4) ? 16 : 0) + half * 8;
    a[2 * v]     = f(row, kOff);
    a[2 * v + 1] = f(row, kOff + 1);
  }
  return a;
}
// D tile: VGPR r -> (M = r + (lane>>4)*8, N = lane&15)

// ---------------- Generic linear: Y[P,Co] = act(X[P,Ci] @ W^T + b) ---------
// Weights staged pre-swizzled (fragment order) + zero-padded in LDS.
// Clamped-index loads (no predication). 4 waves / block, 16 rows each.

template <int CI, int CO, int LDX, int ACT, bool POS, bool F16OUT = false>
__global__ void k_linear(const float* __restrict__ X,
                         const float* __restrict__ W,   // (CO, CI) row-major
                         const float* __restrict__ Bv,  // (CO)
                         const float* __restrict__ pos, // (1024,32) or null
                         float* __restrict__ Y, int ldy) {
  constexpr int KT = ((CI + 31) & ~31) / 32;
  constexpr int NT = (CO + 15) / 16;
  __shared__ alignas(32) _Float16 wsz[KT][NT][512];

  int tid  = threadIdx.x;
  int wave = tid >> 5;
  int lane = tid & 31;
  int col = lane & 15, half = lane >> 4;

#pragma unroll
  for (int kt = 0; kt < KT; ++kt)
#pragma unroll
    for (int t = 0; t < NT; ++t)
#pragma unroll
      for (int i0 = 0; i0 < 512; i0 += 128) {
        int i = i0 + tid;
        int l = i >> 4, e = i & 15;
        int n = t * 16 + (l & 15);
        int k = kt * 32 + frag_k_of(l, e);
        int ni = (n < CO) ? n : (CO - 1);
        int ki = (k < CI) ? k : (CI - 1);
        _Float16 w = (_Float16)W[(size_t)ni * CI + ki];
        wsz[kt][t][i] = (n < CO && k < CI) ? w : (_Float16)0.f;
      }
  __syncthreads();

  int mm = blockIdx.x * 64 + wave * 16;
  v8f acc[NT];
#pragma unroll
  for (int t = 0; t < NT; ++t) acc[t] = vzero();

#pragma unroll
  for (int kt = 0; kt < KT; ++kt) {
    v16h a = load_a_frag([&](int r, int k) -> _Float16 {
      return (_Float16)X[(size_t)(mm + r) * LDX + kt * 32 + k];
    }, lane);
#pragma unroll
    for (int t = 0; t < NT; ++t)
      acc[t] = wmma_f16(a, load_frag(&wsz[kt][t][0], lane), acc[t]);
  }
#pragma unroll
  for (int t = 0; t < NT; ++t) {
    int n = t * 16 + col;
    if (n < CO) {
      float bias = Bv[n];
#pragma unroll
      for (int r = 0; r < 8; ++r) {
        int m = mm + r + half * 8;
        float v = acc[t][r] + bias;
        if (POS) v += pos[(size_t)(m & (NPB - 1)) * 32 + n];
        if (ACT == ACT_RELU) v = fmaxf(v, 0.f);
        else if (ACT == ACT_SIG) v = 1.f / (1.f + __expf(-v));
        if (F16OUT) ((_Float16*)Y)[(size_t)m * ldy + n] = (_Float16)v;
        else        Y[(size_t)m * ldy + n] = v;
      }
    }
  }
}

// ---------------- Attention: one wave per (n, head); S=B=32 sequence -------
// qkv is f16; Q/K^T fragments gathered straight into LDS by the async engine.

__global__ void k_attn(const _Float16* __restrict__ qkv,  // P x 96 (f16)
                       float* __restrict__ o) {           // P x 32
  __shared__ alignas(32) _Float16 aq[2][512];     // Q A-frags (2 M tiles)
  __shared__ alignas(32) _Float16 kb[2][512];     // K^T B-frags (2 N tiles)
  __shared__ alignas(32) _Float16 vb[512];        // V B-frag
  __shared__ alignas(32) _Float16 attsw[2][512];  // softmax(att) A-frags
  __shared__ float attS[32][33];

  int nb = blockIdx.x;          // 0..1023
  int h  = blockIdx.y;          // 0..1
  int lane = threadIdx.x;
  int col = lane & 15, half = lane >> 4;
  const int qo = h * 16, ko = 32 + h * 16, vo = 64 + h * 16;

  // zero the padded upper half (k >= 16 <=> element >= 8) of each fragment
  {
    v8h z = {};
#pragma unroll
    for (int mt = 0; mt < 2; ++mt) {
      *(v8h*)(&aq[mt][lane * 16 + 8]) = z;
      *(v8h*)(&kb[mt][lane * 16 + 8]) = z;
    }
  }
  // async gather of the valid pairs: elements (2v, 2v+1) <-> k = 2v + half*8,
  // which is 4B-contiguous in the f16 source row.
  {
    int row = lane & 15;
#pragma unroll
    for (int mt = 0; mt < 2; ++mt) {
      const _Float16* qsrc =
          qkv + (size_t)((mt * 16 + row) * NPB + nb) * 96 + qo + half * 8;
      const _Float16* ksrc =
          qkv + (size_t)((mt * 16 + row) * NPB + nb) * 96 + ko + half * 8;
#pragma unroll
      for (int v = 0; v < 4; ++v) {
        async_copy_b32(&aq[mt][lane * 16 + 2 * v], qsrc + 2 * v);
        async_copy_b32(&kb[mt][lane * 16 + 2 * v], ksrc + 2 * v);
      }
    }
  }
  // V fragment: pairs run along t (strided) -> stage with normal loads
#pragma unroll
  for (int i0 = 0; i0 < 512; i0 += 32) {
    int i = i0 + lane;
    int l = i >> 4, e = i & 15;
    int n = l & 15;
    int k = frag_k_of(l, e);   // k = t, all 32 valid
    vb[i] = qkv[(size_t)(k * NPB + nb) * 96 + vo + n];
  }
  async_wait0();

  // QK^T / 4
#pragma unroll
  for (int mt = 0; mt < 2; ++mt) {
    v16h a = load_frag(&aq[mt][0], lane);
#pragma unroll
    for (int nt = 0; nt < 2; ++nt) {
      v8f d = wmma_f16(a, load_frag(&kb[nt][0], lane), vzero());
#pragma unroll
      for (int r = 0; r < 8; ++r)
        attS[mt * 16 + r + half * 8][nt * 16 + col] = d[r] * 0.25f;
    }
  }

  // softmax over t (lane L owns row s=L); write probs in fragment order
  {
    float mx = -1e30f;
#pragma unroll
    for (int t = 0; t < 32; ++t) mx = fmaxf(mx, attS[lane][t]);
    float sum = 0.f;
#pragma unroll
    for (int t = 0; t < 32; ++t) {
      float ev = __expf(attS[lane][t] - mx);
      attS[lane][t] = ev;
      sum += ev;
    }
    float inv = 1.f / sum;
    int mt = lane >> 4, row = lane & 15;
#pragma unroll
    for (int t = 0; t < 32; ++t) {
      int khalf = (t >> 3) & 1;
      int dl = row + 16 * khalf;
      attsw[mt][dl * 16 + frag_elem(t)] = (_Float16)(attS[lane][t] * inv);
    }
  }

  // o = att @ V
#pragma unroll
  for (int mt = 0; mt < 2; ++mt) {
    v8f d = wmma_f16(load_frag(&attsw[mt][0], lane), load_frag(vb, lane), vzero());
#pragma unroll
    for (int r = 0; r < 8; ++r) {
      int s = mt * 16 + r + half * 8;
      o[(size_t)(s * NPB + nb) * 32 + h * 16 + col] = d[r];
    }
  }
}

// ---------------- FFN weight pre-swizzle (once per launch) -----------------

__global__ void k_prep_ffw(const float* __restrict__ W1, const float* __restrict__ W2,
                           _Float16* __restrict__ ff1s, _Float16* __restrict__ ff2s) {
  int i = blockIdx.x * 256 + threadIdx.x;   // 0 .. 64*2*512-1
  int tile = i >> 9;
  int idx  = i & 511;
  int l = idx >> 4, e = idx & 15;
  int n = l & 15;
  int k = frag_k_of(l, e);
  int j = tile >> 1, nt = tile & 1;
  ff1s[i] = (_Float16)W1[(size_t)(j * 32 + nt * 16 + n) * 32 + k];
  ff2s[i] = (_Float16)W2[(size_t)(nt * 16 + n) * 2048 + j * 32 + k];
}

// ---------------- Fused FFN: Y = ff2(relu(ff1(X))), hidden stays in LDS ----

__global__ void k_ffn(const float* __restrict__ X,        // P x 32
                      const _Float16* __restrict__ ff1s,  // swizzled W1
                      const float* __restrict__ B1,       // 2048
                      const _Float16* __restrict__ ff2s,  // swizzled W2
                      const float* __restrict__ B2,       // 32
                      float* __restrict__ Y) {            // P x 32
  __shared__ alignas(32) _Float16 aswz[512];   // relu(ff1) tile, fragment order
  int lane = threadIdx.x;
  int col = lane & 15, half = lane >> 4;
  int m0 = blockIdx.x * 16;

  v16h ax = load_a_frag([&](int r, int k) -> _Float16 {
    return (_Float16)X[(size_t)(m0 + r) * 32 + k];
  }, lane);

  v8f acc0 = vzero(), acc1 = vzero();
  for (int j = 0; j < 64; ++j) {          // DFF=2048 in chunks of 32
#pragma unroll
    for (int nt = 0; nt < 2; ++nt) {
      v16h b = load_frag(ff1s + (size_t)(j * 2 + nt) * 512, lane);
      v8f d = wmma_f16(ax, b, vzero());
      float bias = B1[j * 32 + nt * 16 + col];
      int kk = nt * 16 + col;
      int khalf = (kk >> 3) & 1;
      int e = frag_elem(kk);
#pragma unroll
      for (int r = 0; r < 8; ++r) {
        int dl = (r + half * 8) + 16 * khalf;
        aswz[dl * 16 + e] = (_Float16)fmaxf(d[r] + bias, 0.f);
      }
    }
    v16h ah = load_frag(aswz, lane);
    acc0 = wmma_f16(ah, load_frag(ff2s + (size_t)(j * 2 + 0) * 512, lane), acc0);
    acc1 = wmma_f16(ah, load_frag(ff2s + (size_t)(j * 2 + 1) * 512, lane), acc1);
  }
#pragma unroll
  for (int r = 0; r < 8; ++r) {
    int m = m0 + r + half * 8;
    Y[(size_t)m * 32 + col]      = acc0[r] + B2[col];
    Y[(size_t)m * 32 + 16 + col] = acc1[r] + B2[16 + col];
  }
}

// ---------------- Residual + LayerNorm (E=32, one wave32 per row) ----------

__global__ void k_add_ln(const float* __restrict__ A, const float* __restrict__ R,
                         const float* __restrict__ g, const float* __restrict__ be,
                         float* __restrict__ Y, int P) {
  int wid  = (int)((blockIdx.x * blockDim.x + threadIdx.x) >> 5);
  int lane = threadIdx.x & 31;
  if (wid >= P) return;
  float v = A[(size_t)wid * 32 + lane] + R[(size_t)wid * 32 + lane];
  float s = v;
#pragma unroll
  for (int m = 16; m > 0; m >>= 1) s += __shfl_xor(s, m, 32);
  float mean = s * (1.f / 32.f);
  float d = v - mean;
  float vs = d * d;
#pragma unroll
  for (int m = 16; m > 0; m >>= 1) vs += __shfl_xor(vs, m, 32);
  float var = vs * (1.f / 32.f);
  Y[(size_t)wid * 32 + lane] = d * rsqrtf(var + 1e-5f) * g[lane] + be[lane];
}

// ---------------- Noise gate / pad / pack ----------------------------------

__global__ void k_gate(const float* __restrict__ H, const float* __restrict__ Wg,
                       float* __restrict__ Y, int n) {
  int i = blockIdx.x * blockDim.x + threadIdx.x;
  if (i < n) Y[i] = H[i] * Wg[i >> 5];
}

__global__ void k_pad3(const float* __restrict__ X, float* __restrict__ Y, int n) {
  int i = blockIdx.x * blockDim.x + threadIdx.x;   // over P*32
  if (i < n) {
    int p = i >> 5, c = i & 31;
    Y[i] = (c < 3) ? X[(size_t)p * 3 + c] : 0.f;
  }
}

__global__ void k_pack(const float* __restrict__ H, float* __restrict__ O, int n) {
  int i = blockIdx.x * blockDim.x + threadIdx.x;
  if (i < n) {
    int p = i / 60, c = i - p * 60;
    O[i] = H[(size_t)p * 64 + c];
  }
}

// ---------------- kNN: squared norms + fused distance/top-17 ---------------

__global__ void k_sq(const float* __restrict__ F, float* __restrict__ sq, int P) {
  int p = blockIdx.x * blockDim.x + threadIdx.x;
  if (p >= P) return;
  float s = 0.f;
#pragma unroll
  for (int c = 0; c < 24; ++c) {
    float v = F[(size_t)p * 32 + c];
    s += v * v;
  }
  sq[p] = s;
}

__global__ void k_knn(const float* __restrict__ F, const float* __restrict__ sq,
                      int* __restrict__ idx, int P) {
  int p = blockIdx.x * blockDim.x + threadIdx.x;
  if (p >= P) return;
  int base = p & ~(NPB - 1);
  float f[24];
#pragma unroll
  for (int c = 0; c < 24; ++c) f[c] = F[(size_t)p * 32 + c];
  float sp = sq[p];
  float bd[17];
  int   bi[17];
#pragma unroll
  for (int j = 0; j < 17; ++j) { bd[j] = 3.4e38f; bi[j] = 0; }
  for (int m = 0; m < NPB; ++m) {
    int mp = (m + 8) & (NPB - 1);            // prefetch ahead
    __builtin_prefetch(&F[(size_t)(base + mp) * 32], 0, 1);
    const float* g = &F[(size_t)(base + m) * 32];
    float dot = 0.f;
#pragma unroll
    for (int c = 0; c < 24; ++c) dot += f[c] * g[c];
    float d = sp + sq[base + m] - 2.f * dot;
    if (d < bd[16]) {
      int j = 16;
      while (j > 0 && bd[j - 1] > d) { bd[j] = bd[j - 1]; bi[j] = bi[j - 1]; --j; }
      bd[j] = d; bi[j] = base + m;
    }
  }
#pragma unroll
  for (int j = 0; j < 16; ++j) idx[(size_t)p * 16 + j] = bi[j + 1];  // drop self
}

// ---------------- Dense edge conv: one wave per point, M = 16 neighbors ----

template <int CIF, bool REL>
__global__ void k_edge(const float* __restrict__ F,   // P x 32 (24 valid)
                       const int* __restrict__ idx,   // P x 16 (global rows)
                       const float* __restrict__ Wf, const float* __restrict__ bf,
                       const float* __restrict__ Wm, const float* __restrict__ bm,
                       const float* __restrict__ Wl, const float* __restrict__ bl,
                       float* __restrict__ Out, int ldo) {
  constexpr int KTF = ((CIF + 31) & ~31) / 32;       // 1 or 3
  __shared__ alignas(32) _Float16 eg[16][96];        // edge features (A of first)
  __shared__ alignas(32) _Float16 amid[16][64];      // y1 = [relu(first), x_t]
  __shared__ alignas(32) _Float16 alast[16][64];     // y2 = [relu(mid), relu(first), x_t]
  __shared__ alignas(32) _Float16 wfs[KTF][512];     // swizzled weights
  __shared__ alignas(32) _Float16 wms[2][512];
  __shared__ alignas(32) _Float16 wls[2][512];
  __shared__ float    xt[24];
  __shared__ _Float16 xt16[24];
  __shared__ float    red[16][17];

  int p = blockIdx.x;
  int lane = threadIdx.x;
  int col = lane & 15, half = lane >> 4;

  // zero assembled A operands
#pragma unroll
  for (int i0 = 0; i0 < 16 * 96; i0 += 32) (&eg[0][0])[i0 + lane]    = (_Float16)0.f;
#pragma unroll
  for (int i0 = 0; i0 < 16 * 64; i0 += 32) (&amid[0][0])[i0 + lane]  = (_Float16)0.f;
#pragma unroll
  for (int i0 = 0; i0 < 16 * 64; i0 += 32) (&alast[0][0])[i0 + lane] = (_Float16)0.f;

  // stage weights pre-swizzled, zero-padded via clamp+select
#pragma unroll
  for (int kt = 0; kt < KTF; ++kt)
#pragma unroll
    for (int i0 = 0; i0 < 512; i0 += 32) {
      int i = i0 + lane;
      int l = i >> 4, e = i & 15;
      int n = l & 15;
      int k = kt * 32 + frag_k_of(l, e);
      int ni = (n < 12) ? n : 11;
      int ki = (k < CIF) ? k : (CIF - 1);
      _Float16 w = (_Float16)Wf[(size_t)ni * CIF + ki];
      wfs[kt][i] = (n < 12 && k < CIF) ? w : (_Float16)0.f;
    }
#pragma unroll
  for (int kt = 0; kt < 2; ++kt)
#pragma unroll
    for (int i0 = 0; i0 < 512; i0 += 32) {
      int i = i0 + lane;
      int l = i >> 4, e = i & 15;
      int n = l & 15;
      int k = kt * 32 + frag_k_of(l, e);
      int ni = (n < 12) ? n : 11;
      int km = (k < 36) ? k : 35;
      int kl = (k < 48) ? k : 47;
      _Float16 wm = (_Float16)Wm[(size_t)ni * 36 + km];
      _Float16 wl = (_Float16)Wl[(size_t)ni * 48 + kl];
      wms[kt][i] = (n < 12 && k < 36) ? wm : (_Float16)0.f;
      wls[kt][i] = (n < 12 && k < 48) ? wl : (_Float16)0.f;
    }

  if (lane < 24) {
    float v = F[(size_t)p * 32 + lane];
    xt[lane] = v; xt16[lane] = (_Float16)v;
  }
  // broadcast x_t into the concatenated A tiles
#pragma unroll
  for (int i0 = 0; i0 < 16 * 24; i0 += 32) {
    int i = i0 + lane;
    int rr = i / 24, cc = i - rr * 24;
    _Float16 v = xt16[cc];
    amid[rr][12 + cc]  = v;
    alast[rr][24 + cc] = v;
  }

  // edge features: lanes (col, half) -> neighbor col, channels [half*12, +12)
  int nbr = idx[(size_t)p * 16 + col];
#pragma unroll
  for (int cc = 0; cc < 12; ++cc) {
    int c = half * 12 + cc;
    float kf = F[(size_t)nbr * 32 + c];
    float xc = F[(size_t)p * 32 + c];
    if (REL) {
      eg[col][c] = (_Float16)(kf - xc);
    } else {
      eg[col][c]      = (_Float16)xc;
      eg[col][24 + c] = (_Float16)kf;
      eg[col][48 + c] = (_Float16)(kf - xc);
    }
  }

  // first: (16 x CIF) @ (CIF x 12) -> relu -> amid[:,0:12], alast[:,12:24]
  v8f accf = vzero();
#pragma unroll
  for (int kt = 0; kt < KTF; ++kt) {
    v16h a = load_a_frag([&](int r, int k) -> _Float16 {
      return eg[r][kt * 32 + k];
    }, lane);
    accf = wmma_f16(a, load_frag(&wfs[kt][0], lane), accf);
  }
  if (col < 12) {
    float bias = bf[col];
#pragma unroll
    for (int r = 0; r < 8; ++r) {
      int m = r + half * 8;
      _Float16 v = (_Float16)fmaxf(accf[r] + bias, 0.f);
      amid[m][col]       = v;
      alast[m][12 + col] = v;
    }
  }

  // mid: (16 x 36) @ (36 x 12) -> relu -> alast[:,0:12]
  v8f accm = vzero();
#pragma unroll
  for (int kt = 0; kt < 2; ++kt) {
    v16h a = load_a_frag([&](int r, int k) -> _Float16 {
      return amid[r][kt * 32 + k];
    }, lane);
    accm = wmma_f16(a, load_frag(&wms[kt][0], lane), accm);
  }
  if (col < 12) {
    float bias = bm[col];
#pragma unroll
    for (int r = 0; r < 8; ++r)
      alast[r + half * 8][col] = (_Float16)fmaxf(accm[r] + bias, 0.f);
  }

  // last: (16 x 48) @ (48 x 12), no relu
  v8f accl = vzero();
#pragma unroll
  for (int kt = 0; kt < 2; ++kt) {
    v16h a = load_a_frag([&](int r, int k) -> _Float16 {
      return alast[r][kt * 32 + k];
    }, lane);
    accl = wmma_f16(a, load_frag(&wls[kt][0], lane), accl);
  }
  if (col < 12) {
    float bias = bl[col];
#pragma unroll
    for (int r = 0; r < 8; ++r)
      red[r + half * 8][col] = accl[r] + bias;
  }

  // max over the 16 neighbors (M axis); x_t part is neighbor-invariant
  if (lane < 12) {
    float m1 = -1e30f, m2 = -1e30f, m3 = -1e30f;
#pragma unroll
    for (int q = 0; q < 16; ++q) {
      m1 = fmaxf(m1, red[q][lane]);           // last
      m2 = fmaxf(m2, (float)alast[q][lane]);  // relu(mid)
      m3 = fmaxf(m3, (float)amid[q][lane]);   // relu(first)
    }
    Out[(size_t)p * ldo + lane]      = m1;
    Out[(size_t)p * ldo + 12 + lane] = m2;
    Out[(size_t)p * ldo + 24 + lane] = m3;
  }
  if (lane < 24) Out[(size_t)p * ldo + 36 + lane] = xt[lane];
}

// ---------------- Host orchestration ---------------------------------------

extern "C" void kernel_launch(void* const* d_in, const int* in_sizes, int n_in,
                              void* d_out, int out_size, void* d_ws, size_t ws_size,
                              hipStream_t stream) {
  (void)in_sizes; (void)n_in; (void)out_size; (void)ws_size;
  // pytree order (sorted dict keys): x, convs[i]{first,last,mid}{b,w}, nrt{...}, trans[i]{b,w}
  const float* x_in = (const float*)d_in[0];
  auto cF_b = [&](int i) { return (const float*)d_in[1 + i * 6 + 0]; };
  auto cF_w = [&](int i) { return (const float*)d_in[1 + i * 6 + 1]; };
  auto cL_b = [&](int i) { return (const float*)d_in[1 + i * 6 + 2]; };
  auto cL_w = [&](int i) { return (const float*)d_in[1 + i * 6 + 3]; };
  auto cM_b = [&](int i) { return (const float*)d_in[1 + i * 6 + 4]; };
  auto cM_w = [&](int i) { return (const float*)d_in[1 + i * 6 + 5]; };
  const float* fc_in_b    = (const float*)d_in[25];
  const float* fc_in_w    = (const float*)d_in[26];
  const float* fc_out_b   = (const float*)d_in[27];
  const float* fc_out_w   = (const float*)d_in[28];
  const float* ff1_b      = (const float*)d_in[29];
  const float* ff1_w      = (const float*)d_in[30];
  const float* ff2_b      = (const float*)d_in[31];
  const float* ff2_w      = (const float*)d_in[32];
  const float* in_proj_b  = (const float*)d_in[33];
  const float* in_proj_w  = (const float*)d_in[34];
  const float* ln1_b      = (const float*)d_in[35];
  const float* ln1_g      = (const float*)d_in[36];
  const float* ln2_b      = (const float*)d_in[37];
  const float* ln2_g      = (const float*)d_in[38];
  const float* na1_b      = (const float*)d_in[39];
  const float* na1_w      = (const float*)d_in[40];
  const float* na2_b      = (const float*)d_in[41];
  const float* na2_w      = (const float*)d_in[42];
  const float* out_proj_b = (const float*)d_in[43];
  const float* out_proj_w = (const float*)d_in[44];
  const float* pos        = (const float*)d_in[45];
  auto tr_b = [&](int i) { return (const float*)d_in[46 + i * 2 + 0]; };
  auto tr_w = [&](int i) { return (const float*)d_in[46 + i * 2 + 1]; };

  // workspace: f16 regions first (32B aligned), then float buffers
  _Float16* ff1s   = (_Float16*)d_ws;               // 64*2*512 f16
  _Float16* ff2s   = ff1s + 64 * 2 * 512;
  _Float16* qkv16  = ff2s + 64 * 2 * 512;           // P x 96 f16
  float* ws = (float*)(qkv16 + (size_t)PTS * 96);
  float* buf_h   = ws;  ws += (size_t)PTS * 64;
  float* buf_x0  = ws;  ws += (size_t)PTS * 32;
  float* buf_t   = ws;  ws += (size_t)PTS * 32;
  float* buf_x   = ws;  ws += (size_t)PTS * 32;
  float* buf_o   = ws;  ws += (size_t)PTS * 32;
  float* buf_tmp = ws;  ws += (size_t)PTS * 32;
  float* buf_x2  = ws;  ws += (size_t)PTS * 32;
  float* buf_he  = ws;  ws += (size_t)PTS * 32;
  float* buf_na  = ws;  ws += (size_t)PTS * 32;
  float* buf_w   = ws;  ws += (size_t)PTS;
  float* buf_hw  = ws;  ws += (size_t)PTS * 32;
  float* buf_nt  = ws;  ws += (size_t)PTS * 32;
  float* buf_sq  = ws;  ws += (size_t)PTS;
  int*   buf_idx = (int*)ws;

  const dim3 linG(PTS / 64), linB(128);
  k_prep_ffw<<<(64 * 2 * 512) / 256, 256, 0, stream>>>(ff1_w, ff2_w, ff1s, ff2s);
  k_pad3<<<(PTS * 32 + 255) / 256, 256, 0, stream>>>(x_in, buf_x0, PTS * 32);

  for (int i = 0; i < 4; ++i) {
    // h = trans(h); relu for i > 0
    if (i == 0)
      k_linear<3, 24, 32, ACT_NONE, false><<<linG, linB, 0, stream>>>(
          buf_x0, tr_w(0), tr_b(0), nullptr, buf_t, 32);
    else
      k_linear<60, 24, 64, ACT_RELU, false><<<linG, linB, 0, stream>>>(
          buf_h, tr_w(i), tr_b(i), nullptr, buf_t, 32);
    // --- noise transformer ---
    k_linear<24, 32, 32, ACT_NONE, true><<<linG, linB, 0, stream>>>(
        buf_t, fc_in_w, fc_in_b, pos, buf_x, 32);
    k_linear<32, 96, 32, ACT_NONE, false, true><<<linG, linB, 0, stream>>>(
        buf_x, in_proj_w, in_proj_b, nullptr, (float*)qkv16, 96);
    k_attn<<<dim3(NPB, 2), dim3(32), 0, stream>>>(qkv16, buf_o);
    k_linear<32, 32, 32, ACT_NONE, false><<<linG, linB, 0, stream>>>(
        buf_o, out_proj_w, out_proj_b, nullptr, buf_tmp, 32);
    k_add_ln<<<PTS / 8, 256, 0, stream>>>(buf_x, buf_tmp, ln1_g, ln1_b, buf_x2, PTS);
    k_ffn<<<PTS / 16, 32, 0, stream>>>(buf_x2, ff1s, ff1_b, ff2s, ff2_b, buf_tmp);
    k_add_ln<<<PTS / 8, 256, 0, stream>>>(buf_x2, buf_tmp, ln2_g, ln2_b, buf_he, PTS);
    k_linear<32, 16, 32, ACT_RELU, false><<<linG, linB, 0, stream>>>(
        buf_he, na1_w, na1_b, nullptr, buf_na, 32);
    k_linear<16, 1, 32, ACT_SIG, false><<<linG, linB, 0, stream>>>(
        buf_na, na2_w, na2_b, nullptr, buf_w, 1);
    k_gate<<<(PTS * 32 + 255) / 256, 256, 0, stream>>>(buf_he, buf_w, buf_hw, PTS * 32);
    k_linear<32, 24, 32, ACT_NONE, false><<<linG, linB, 0, stream>>>(
        buf_hw, fc_out_w, fc_out_b, nullptr, buf_nt, 32);
    // --- dense edge conv ---
    k_sq<<<(PTS + 255) / 256, 256, 0, stream>>>(buf_nt, buf_sq, PTS);
    k_knn<<<PTS / 256, 256, 0, stream>>>(buf_nt, buf_sq, buf_idx, PTS);
    if (i == 0)
      k_edge<24, true><<<PTS, dim3(32), 0, stream>>>(
          buf_nt, buf_idx, cF_w(0), cF_b(0), cM_w(0), cM_b(0), cL_w(0), cL_b(0),
          buf_h, 64);
    else
      k_edge<72, false><<<PTS, dim3(32), 0, stream>>>(
          buf_nt, buf_idx, cF_w(i), cF_b(i), cM_w(i), cM_b(i), cL_w(i), cL_b(i),
          buf_h, 64);
  }
  k_pack<<<(PTS * 60 + 255) / 256, 256, 0, stream>>>(buf_h, (float*)d_out, PTS * 60);
}
